// CALIP_PF_Block_76373108457749
// MI455X (gfx1250) — matrix operand order repaired and checked by reading, not compiled, not run on hardware
//
#include <hip/hip_runtime.h>

#define B 128
#define M 196
#define MP 208   // M padded to 13*16
#define MT 13
#define D 512
#define K 1000
#define KP 1024  // K padded to 64*16

typedef __attribute__((ext_vector_type(16))) _Float16 v16h;
typedef __attribute__((ext_vector_type(8)))  float    v8f;

union Frag { v16h v; uint4 q[2]; };

__device__ __forceinline__ v8f wmma_f16(const Frag& a, const Frag& b, v8f c) {
  return __builtin_amdgcn_wmma_f32_16x16x32_f16(false, a.v, false, b.v, (short)0, c, false, false);
}

// ---------------------------------------------------------------------------
// Kernel 1: normalize Fs rows (fp32), write fp16 Fs_norm (padded to MP rows),
//           and v[b][m] = <Fv_b, Fs_norm[b,m]> in fp32.
// grid (MP, B), block 128
__global__ void k_norm(const float* __restrict__ Fs, const float* __restrict__ Fv,
                       _Float16* __restrict__ fsn, float* __restrict__ vbuf) {
  int m = blockIdx.x, b = blockIdx.y, tid = threadIdx.x;
  _Float16* dst = fsn + ((size_t)(b * MP + m)) * D;
  if (m >= M) {
    for (int j = 0; j < 4; ++j) dst[tid * 4 + j] = (_Float16)0.f;
    if (tid == 0) vbuf[b * MP + m] = 0.f;
    return;
  }
  __shared__ float red[4], red2[4];
  const float* src = Fs + ((size_t)(b * M + m)) * D;
  float x[4]; float ss = 0.f;
  for (int j = 0; j < 4; ++j) { x[j] = src[tid * 4 + j]; ss += x[j] * x[j]; }
  for (int s = 1; s < 32; s <<= 1) ss += __shfl_xor(ss, s);
  int w = tid >> 5, lane = tid & 31;
  if (lane == 0) red[w] = ss;
  __syncthreads();
  ss = red[0] + red[1] + red[2] + red[3];
  float rn = rsqrtf(ss);
  const float* fvb = Fv + (size_t)b * D;
  float dv = 0.f;
  for (int j = 0; j < 4; ++j) {
    float xn = x[j] * rn;
    dst[tid * 4 + j] = (_Float16)xn;
    dv += fvb[tid * 4 + j] * xn;
  }
  for (int s = 1; s < 32; s <<= 1) dv += __shfl_xor(dv, s);
  if (lane == 0) red2[w] = dv;
  __syncthreads();
  if (tid == 0) vbuf[b * MP + m] = red2[0] + red2[1] + red2[2] + red2[3];
}

// ---------------------------------------------------------------------------
// Kernel 2a: Ft fp32 -> fp16, zero-padded to KP rows.  grid KP*D/256, block 256
__global__ void k_ftcvt(const float* __restrict__ Ft, _Float16* __restrict__ ft16) {
  int idx = blockIdx.x * 256 + threadIdx.x;
  int k = idx >> 9;
  int d = idx & 511;
  ft16[idx] = (k < K) ? (_Float16)Ft[(size_t)k * D + d] : (_Float16)0.f;
}

// Kernel 2b: pre-pack Ft (K x D) into B-fragment layout for P*Ft matmul.
// pack[kc][dt][lane][i] = Ft16[kc*32 + (lane>>4)*16 + i][dt*16 + (lane&15)]
// grid 32*32, block 32
__global__ void k_pack(const _Float16* __restrict__ ft16, _Float16* __restrict__ pack) {
  int bx = blockIdx.x; int kc = bx >> 5, dt = bx & 31;
  int lane = threadIdx.x, lo = lane & 15, hi = lane >> 4;
  union { _Float16 h[16]; uint4 q[2]; } t;
  #pragma unroll
  for (int i = 0; i < 16; ++i) {
    int k = kc * 32 + hi * 16 + i;
    t.h[i] = ft16[(size_t)k * D + dt * 16 + lo];
  }
  uint4* dst = (uint4*)(pack + ((size_t)((kc * 32 + dt) * 32 + lane)) * 16);
  dst[0] = t.q[0]; dst[1] = t.q[1];
}

// ---------------------------------------------------------------------------
// Kernel 3: per (b, mtile): A row-tile (16 x KP) via WMMA -> softmax over k
// (in-place fp16 A -> P in LDS) -> Fsa tile = P*Ft via WMMA -> partial
// mean/max over valid m written to psum/pmax.
// grid (MT, B), block 128 (4 waves)
__global__ void k_row(const _Float16* __restrict__ fsn, const _Float16* __restrict__ ft16,
                      const _Float16* __restrict__ pack,
                      float* __restrict__ psum, float* __restrict__ pmax) {
  int mt = blockIdx.x, b = blockIdx.y, tid = threadIdx.x;
  int w = tid >> 5, lane = tid & 31, lo = lane & 15, hi = lane >> 4;
  __shared__ _Float16 P[16][KP];   // A tile fp16, overwritten in-place by exp()
  __shared__ float rrec[16];

  const _Float16* arow = fsn + ((size_t)(b * MP + mt * 16 + lo)) * D;
  // Phase 1: A = Fs_norm * Ft^T ; wave w owns ktiles [w*16, w*16+16), 4 at a time.
  for (int jg = 0; jg < 4; ++jg) {
    v8f acc[4] = {};
    #pragma unroll 1
    for (int dc = 0; dc < 16; ++dc) {
      Frag a;
      const uint4* pa = (const uint4*)(arow + dc * 32 + hi * 16);
      a.q[0] = pa[0]; a.q[1] = pa[1];
      #pragma unroll
      for (int u = 0; u < 4; ++u) {
        int kt = w * 16 + jg * 4 + u;
        Frag bb;
        const uint4* pb = (const uint4*)(ft16 + (size_t)(kt * 16 + lo) * D + dc * 32 + hi * 16);
        bb.q[0] = pb[0]; bb.q[1] = pb[1];
        acc[u] = wmma_f16(a, bb, acc[u]);
      }
    }
    #pragma unroll
    for (int u = 0; u < 4; ++u) {
      int kcol = (w * 16 + jg * 4 + u) * 16 + lo;
      #pragma unroll
      for (int r = 0; r < 8; ++r) P[r + 8 * hi][kcol] = (_Float16)acc[u][r];
    }
  }
  __syncthreads();

  // Phase 2: softmax over k per row (8 threads per row), P <- exp(2(a-max))
  {
    int row = tid >> 3, g = tid & 7;
    float mx = -3.0e38f;
    for (int k = g; k < K; k += 8) mx = fmaxf(mx, (float)P[row][k]);
    mx = fmaxf(mx, __shfl_xor(mx, 1));
    mx = fmaxf(mx, __shfl_xor(mx, 2));
    mx = fmaxf(mx, __shfl_xor(mx, 4));
    float s = 0.f;
    for (int k = g; k < KP; k += 8) {
      float e = 0.f;
      if (k < K) { e = __expf(2.f * ((float)P[row][k] - mx)); s += e; }
      P[row][k] = (_Float16)e;
    }
    s += __shfl_xor(s, 1); s += __shfl_xor(s, 2); s += __shfl_xor(s, 4);
    if (g == 0) rrec[row] = 1.f / s;
  }
  __syncthreads();

  // Phase 3: Fsa tile = P * Ft (pre-packed B), fold 1/rowsum, reduce over m.
  // Wave w owns dtiles [w*8, w*8+8), 4 at a time.
  for (int tg = 0; tg < 2; ++tg) {
    v8f acc[4] = {};
    #pragma unroll 1
    for (int kc = 0; kc < 32; ++kc) {
      Frag a;
      const uint4* pa = (const uint4*)&P[lo][kc * 32 + hi * 16];
      a.q[0] = pa[0]; a.q[1] = pa[1];
      #pragma unroll
      for (int u = 0; u < 4; ++u) {
        int dt = w * 8 + tg * 4 + u;
        Frag bb;
        const uint4* pb = (const uint4*)(pack + ((size_t)((kc * 32 + dt) * 32 + lane)) * 16);
        bb.q[0] = pb[0]; bb.q[1] = pb[1];
        acc[u] = wmma_f16(a, bb, acc[u]);
      }
    }
    #pragma unroll
    for (int u = 0; u < 4; ++u) {
      int dt = w * 8 + tg * 4 + u;
      float ps = 0.f, pm = -3.0e38f;
      #pragma unroll
      for (int r = 0; r < 8; ++r) {
        int ml = r + 8 * hi;
        float f = acc[u][r] * rrec[ml];
        if (mt * 16 + ml < M) { ps += f; pm = fmaxf(pm, f); }
      }
      ps += __shfl_xor(ps, 16);
      pm = fmaxf(pm, __shfl_xor(pm, 16));
      if (lane < 16) {
        int off = (b * MT + mt) * D + dt * 16 + lane;
        psum[off] = ps; pmax[off] = pm;
      }
    }
  }
}

// ---------------------------------------------------------------------------
// Kernel 4: Fva[b][d] = mean_m Fsa + max_m Fsa from per-mtile partials.
// grid B, block D
__global__ void k_fva(const float* __restrict__ psum, const float* __restrict__ pmax,
                      float* __restrict__ fva) {
  int b = blockIdx.x, d = threadIdx.x;
  float s = 0.f, m = -3.0e38f;
  for (int t = 0; t < MT; ++t) {
    int off = (b * MT + t) * D + d;
    s += psum[off];
    m = fmaxf(m, pmax[off]);
  }
  fva[b * D + d] = s * (1.f / 196.f) + m;
}

// ---------------------------------------------------------------------------
// k_col helper: N consecutive mtiles against this wave's 16-wide k strip.
// B-frag loaded once per dc, reused across the N mtiles.
template <int N>
__device__ __forceinline__ void col_mats(int mg, int b, int kbase, int lo, int hi, int w,
                                         const _Float16* __restrict__ fsn,
                                         const _Float16* __restrict__ ft16,
                                         float (*Acol)[64]) {
  v8f acc[N] = {};
  const _Float16* bbase = ft16 + (size_t)(kbase + lo) * D + hi * 16;
  #pragma unroll 1
  for (int dc = 0; dc < 16; ++dc) {
    Frag bb;
    const uint4* pb = (const uint4*)(bbase + dc * 32);
    bb.q[0] = pb[0]; bb.q[1] = pb[1];
    #pragma unroll
    for (int u = 0; u < N; ++u) {
      const _Float16* arow = fsn + ((size_t)(b * MP + (mg + u) * 16 + lo)) * D;
      Frag a;
      const uint4* pa = (const uint4*)(arow + dc * 32 + hi * 16);
      a.q[0] = pa[0]; a.q[1] = pa[1];
      acc[u] = wmma_f16(a, bb, acc[u]);
    }
  }
  #pragma unroll
  for (int u = 0; u < N; ++u) {
    #pragma unroll
    for (int r = 0; r < 8; ++r)
      Acol[(mg + u) * 16 + r + 8 * hi][w * 16 + lo] = acc[u][r];
  }
}

// Kernel 5: per (b, 64-wide k block): A col-tile (MP x 64) via WMMA ->
// softmax over m -> term2 = sum p*v ; plus fp32 dots <Fv,Ft_k>, <Fva,Ft_k>;
// combine into output.
// grid (KP/64, B), block 128 (4 waves)
__global__ void k_col(const _Float16* __restrict__ fsn, const _Float16* __restrict__ ft16,
                      const float* __restrict__ Ftf, const float* __restrict__ Fv,
                      const float* __restrict__ fva, const float* __restrict__ vbuf,
                      const float* __restrict__ lsptr, float* __restrict__ out) {
  int kb = blockIdx.x, b = blockIdx.y, tid = threadIdx.x;
  int w = tid >> 5, lane = tid & 31, lo = lane & 15, hi = lane >> 4;
  __shared__ float Acol[MP][64];
  __shared__ float t2buf[64], d1buf[64], d3buf[64];

  int kbase = kb * 64 + w * 16;
  col_mats<4>(0,  b, kbase, lo, hi, w, fsn, ft16, Acol);
  col_mats<4>(4,  b, kbase, lo, hi, w, fsn, ft16, Acol);
  col_mats<4>(8,  b, kbase, lo, hi, w, fsn, ft16, Acol);
  col_mats<1>(12, b, kbase, lo, hi, w, fsn, ft16, Acol);
  __syncthreads();

  float ls = __expf(lsptr[0]);
  if (tid < 64) {
    int c = tid;
    float mx = -3.0e38f;
    for (int m = 0; m < M; ++m) mx = fmaxf(mx, Acol[m][c]);
    float s = 0.f, t2 = 0.f;
    const float* vb = vbuf + b * MP;
    for (int m = 0; m < M; ++m) {
      float e = __expf(2.f * (Acol[m][c] - mx));
      s += e; t2 += e * vb[m];
    }
    t2buf[c] = t2 / s;
  } else {
    int c = tid - 64;
    int k = kb * 64 + c;
    float d1 = 0.f, d3 = 0.f;
    if (k < K) {
      const float4* ftr = (const float4*)(Ftf + (size_t)k * D);
      const float4* fvb = (const float4*)(Fv + (size_t)b * D);
      const float4* fab = (const float4*)(fva + (size_t)b * D);
      for (int d = 0; d < D / 4; ++d) {
        float4 f = ftr[d], a = fvb[d], c4 = fab[d];
        d1 += a.x * f.x + a.y * f.y + a.z * f.z + a.w * f.w;
        d3 += c4.x * f.x + c4.y * f.y + c4.z * f.z + c4.w * f.w;
      }
    }
    d1buf[c] = d1; d3buf[c] = d3;
  }
  __syncthreads();
  if (tid < 64) {
    int k = kb * 64 + tid;
    if (k < K)
      out[(size_t)b * K + k] = ls * (1.0f * d1buf[tid] + 0.1f * t2buf[tid] + 0.1f * d3buf[tid]);
  }
}

// ---------------------------------------------------------------------------
extern "C" void kernel_launch(void* const* d_in, const int* in_sizes, int n_in,
                              void* d_out, int out_size, void* d_ws, size_t ws_size,
                              hipStream_t stream) {
  const float* Fs  = (const float*)d_in[0];
  const float* Ft  = (const float*)d_in[1];
  const float* Fv  = (const float*)d_in[2];
  const float* lsc = (const float*)d_in[3];
  float* out = (float*)d_out;

  char* ws = (char*)d_ws;
  size_t off = 0;
  _Float16* fsn  = (_Float16*)(ws + off); off += (size_t)B * MP * D * 2;        // 27.3 MB
  _Float16* ft16 = (_Float16*)(ws + off); off += (size_t)KP * D * 2;            //  1.0 MB
  _Float16* pack = (_Float16*)(ws + off); off += (size_t)32 * 32 * 32 * 16 * 2; //  1.0 MB
  float* vbuf = (float*)(ws + off); off += (size_t)B * MP * 4;                  //  0.1 MB
  float* psum = (float*)(ws + off); off += (size_t)B * MT * D * 4;              //  3.4 MB
  float* pmax = (float*)(ws + off); off += (size_t)B * MT * D * 4;              //  3.4 MB
  float* fva  = (float*)(ws + off); off += (size_t)B * D * 4;                   //  0.3 MB

  hipLaunchKernelGGL(k_norm,  dim3(MP, B),          dim3(128), 0, stream, Fs, Fv, fsn, vbuf);
  hipLaunchKernelGGL(k_ftcvt, dim3((KP * D) / 256), dim3(256), 0, stream, Ft, ft16);
  hipLaunchKernelGGL(k_pack,  dim3(32 * 32),        dim3(32),  0, stream, ft16, pack);
  hipLaunchKernelGGL(k_row,   dim3(MT, B),          dim3(128), 0, stream, fsn, ft16, pack, psum, pmax);
  hipLaunchKernelGGL(k_fva,   dim3(B),              dim3(D),   0, stream, psum, pmax, fva);
  hipLaunchKernelGGL(k_col,   dim3(KP / 64, B),     dim3(128), 0, stream, fsn, ft16, Ft, Fv, fva, vbuf, lsc, out);
}
